// NGCF_55989193671007
// MI455X (gfx1250) — matrix operand order repaired and checked by reading, not compiled, not run on hardware
//
#include <hip/hip_runtime.h>
#include <hip/hip_bf16.h>
#include <math.h>

// NGCF forward for MI455X (gfx1250, wave32).
//
// Pipeline per layer (x3):
//   1) zero side[N,64]
//   2) edge scatter: side[col] += ego[row] * w   (float4 loads + global_atomic_add_f32)
//   3) WMMA fused GEMM block: ego' = leakyrelu(side@Wgc + b + (ego*side)@Wbi + b)
//      + row L2-normalized copy into concat[N,256] at column (k+1)*64
// Weights are prepacked once per call into bf16 WMMA B-fragment order so the
// hot GEMM kernel loads each fragment as contiguous 32 B per lane.
// Then gather users/pos/neg rows of concat into d_out (each duplicated).

#define N_NODES 100000
#define N_EDGES 1000000
#define DIM     64
#define N_LAYERS 3
#define BATCH   4096

typedef __attribute__((ext_vector_type(16))) __bf16 v16bf;
typedef __attribute__((ext_vector_type(8)))  float  v8f;

// Packed-weight geometry: per layer -> [mat(2)][kstep(2)][ntile(4)][lane(32)][i(16)] bf16
#define PACK_PER_LAYER (2 * 2 * 4 * 32 * 16)   // 8192 bf16 = 16 KB

// ---------------------------------------------------------------- utilities

__global__ void ngcf_zero_kernel(float* __restrict__ p, int n) {
    int gid = blockIdx.x * blockDim.x + threadIdx.x;
    if (gid < n) p[gid] = 0.0f;
}

// concat[:, 0:64] = embeds  (layer-0 slice of the concatenated embedding)
__global__ void ngcf_copy_embeds_kernel(const float* __restrict__ embeds,
                                        float* __restrict__ concat) {
    int gid = blockIdx.x * blockDim.x + threadIdx.x;   // over N*16 float4's
    if (gid >= N_NODES * 16) return;
    int node = gid >> 4;
    int c4   = (gid & 15) << 2;
    const float4 v = *reinterpret_cast<const float4*>(embeds + (size_t)node * DIM + c4);
    *reinterpret_cast<float4*>(concat + (size_t)node * (4 * DIM) + c4) = v;
}

// ----------------------------------------------- weight prepack (f32 -> bf16)
// Rearranges W_gc/W_bi into the per-lane B-fragment order the GEMM kernel reads:
//   B 32x16 bf16, contiguous k order (as A): lanes 0-15 hold K=kb+0..15, lanes 16-31 K=kb+16..31,
//   column N = ntile*16 + (lane&15); element i is K offset within half.
__global__ void ngcf_pack_weights_kernel(const float* __restrict__ Wgc,
                                         const float* __restrict__ Wbi,
                                         __bf16* __restrict__ packed) {
    int t = blockIdx.x * blockDim.x + threadIdx.x;
    if (t >= N_LAYERS * PACK_PER_LAYER) return;
    int i     = t & 15;
    int lane  = (t >> 4) & 31;
    int ntile = (t >> 9) & 3;
    int kbi   = (t >> 11) & 1;
    int mat   = (t >> 12) & 1;
    int layer = t >> 13;
    const float* src = (mat ? Wbi : Wgc) + (size_t)layer * DIM * DIM;
    int K   = kbi * 32 + (lane >> 4) * 16 + i;
    int col = ntile * 16 + (lane & 15);
    packed[t] = (__bf16)src[K * DIM + col];
}

// ------------------------------------------------------- edge scatter-add
// 16 threads per edge, each moves a float4 (4 of the 64 dims).
__global__ void ngcf_scatter_kernel(const float* __restrict__ ego,
                                    const float* __restrict__ ew,
                                    const int* __restrict__ edge_index,
                                    float* __restrict__ side) {
    int gid = blockIdx.x * blockDim.x + threadIdx.x;
    int e = gid >> 4;
    if (e >= N_EDGES) return;
    int off = (gid & 15) << 2;
    int r = edge_index[e];             // row
    int c = edge_index[N_EDGES + e];   // col
    float w = ew[e];
    const float4 v = *reinterpret_cast<const float4*>(ego + (size_t)r * DIM + off);
    float* dst = side + (size_t)c * DIM + off;
    __hip_atomic_fetch_add(dst + 0, v.x * w, __ATOMIC_RELAXED, __HIP_MEMORY_SCOPE_AGENT);
    __hip_atomic_fetch_add(dst + 1, v.y * w, __ATOMIC_RELAXED, __HIP_MEMORY_SCOPE_AGENT);
    __hip_atomic_fetch_add(dst + 2, v.z * w, __ATOMIC_RELAXED, __HIP_MEMORY_SCOPE_AGENT);
    __hip_atomic_fetch_add(dst + 3, v.w * w, __ATOMIC_RELAXED, __HIP_MEMORY_SCOPE_AGENT);
}

// --------------------------------------------------- fused WMMA layer GEMM
// Block = 128 threads = 4 waves; one block computes one 16-row tile of:
//   x   = leakyrelu( side16 @ Wgc + bgc + (ego16*side16) @ Wbi + bbi )
//   ego_out rows    = x            (raw, feeds next layer)
//   concat slice    = x / max(||x||_2, 1e-12)
// Wave w owns output columns [w*16, w*16+16).  Grid = N/16 = 6250 exact,
// so EXEC is all-ones for every WMMA (ISA requirement).
__global__ __launch_bounds__(128) void ngcf_gemm_kernel(
    const float* __restrict__ side, const float* __restrict__ ego,
    const v16bf* __restrict__ packedW,   // this layer's packed weights
    const float* __restrict__ bgc, const float* __restrict__ bbi,
    float* __restrict__ ego_out, float* __restrict__ concat, int layer) {

    __shared__ float s_side[16][68];   // +4 pad keeps float4 alignment, spreads banks
    __shared__ float s_ego [16][68];
    __shared__ float s_sumsq[16];

    const int tid   = threadIdx.x;
    const int rbase = blockIdx.x * 16;

    // Stage the 16x64 side/ego tiles into LDS (float4, fully coalesced).
    for (int q = tid; q < 256; q += 128) {
        int r  = q >> 4;
        int c4 = (q & 15) << 2;
        const float4 sv  = *reinterpret_cast<const float4*>(side + (size_t)(rbase + r) * DIM + c4);
        const float4 evv = *reinterpret_cast<const float4*>(ego + (size_t)(rbase + r) * DIM + c4);
        *reinterpret_cast<float4*>(&s_side[r][c4]) = sv;
        *reinterpret_cast<float4*>(&s_ego [r][c4]) = evv;
    }
    if (tid < 16) s_sumsq[tid] = 0.0f;
    __syncthreads();

    const int lane = tid & 31;
    const int wave = tid >> 5;     // = ntile
    const int m    = lane & 15;    // A row / B column within tile
    const int hi   = lane >> 4;    // half-wave select
    const int col  = wave * 16 + m;  // global output column (0..63)

    v8f acc = {};
    #pragma unroll
    for (int kbi = 0; kbi < 2; ++kbi) {
        const int kb = kbi * 32;
        // ---- A fragments (contiguous k order, same as packed B):
        // lane half hi: K base = kb + hi*16; elems 0..15 -> K+0..15
        v16bf a_s, a_b;
        const int ka = kb + hi * 16;
        #pragma unroll
        for (int i = 0; i < 16; ++i) {
            int K = ka + i;
            float sv = s_side[m][K];
            a_s[i] = (__bf16)sv;
            a_b[i] = (__bf16)(sv * s_ego[m][K]);
        }
        // ---- B fragments: prepacked, contiguous 32 B per lane (2x b128 loads).
        const v16bf b_gc = packedW[((size_t)(0 * 2 + kbi) * 4 + wave) * 32 + lane];
        const v16bf b_bi = packedW[((size_t)(1 * 2 + kbi) * 4 + wave) * 32 + lane];
        acc = __builtin_amdgcn_wmma_f32_16x16x32_bf16(false, a_s, false, b_gc,
                                                      (short)0, acc, false, false);
        acc = __builtin_amdgcn_wmma_f32_16x16x32_bf16(false, a_b, false, b_bi,
                                                      (short)0, acc, false, false);
    }

    // C/D layout: element r -> row M = hi*8 + r, column N = col.
    const float bias = bgc[col] + bbi[col];
    float xs[8];
    #pragma unroll
    for (int r = 0; r < 8; ++r) {
        int M = hi * 8 + r;
        float v = acc[r] + bias;
        float x = (v > 0.0f) ? v : 0.2f * v;           // leaky_relu(0.2)
        xs[r] = x;
        ego_out[(size_t)(rbase + M) * DIM + col] = x;  // raw, next layer input
        atomicAdd(&s_sumsq[M], x * x);                 // ds_add_f32 reduction
    }
    __syncthreads();
    #pragma unroll
    for (int r = 0; r < 8; ++r) {
        int M = hi * 8 + r;
        float nrm = sqrtf(s_sumsq[M]);
        float scale = 1.0f / fmaxf(nrm, 1e-12f);
        concat[(size_t)(rbase + M) * (4 * DIM) + (layer + 1) * DIM + col] = xs[r] * scale;
    }
}

// ------------------------------------------------------------- final gather
// One block per (segment, batch-row); segment 0/1/2 = users/pos/neg; each row
// written twice (outputs 0..2 and 3..5 are identical in the reference tuple).
__global__ __launch_bounds__(64) void ngcf_gather_kernel(
    const float* __restrict__ concat,
    const int* __restrict__ users, const int* __restrict__ pos,
    const int* __restrict__ neg, float* __restrict__ out) {
    int b   = blockIdx.x & (BATCH - 1);
    int seg = blockIdx.x >> 12;
    const int* idxs = (seg == 0) ? users : ((seg == 1) ? pos : neg);
    int node = idxs[b];
    const float4 v = *reinterpret_cast<const float4*>(
        concat + (size_t)node * (4 * DIM) + threadIdx.x * 4);
    size_t o1 = ((size_t)seg * BATCH + b) * (4 * DIM) + threadIdx.x * 4;
    size_t o2 = ((size_t)(seg + 3) * BATCH + b) * (4 * DIM) + threadIdx.x * 4;
    *reinterpret_cast<float4*>(out + o1) = v;
    *reinterpret_cast<float4*>(out + o2) = v;
}

// ------------------------------------------------------------------ launch
extern "C" void kernel_launch(void* const* d_in, const int* in_sizes, int n_in,
                              void* d_out, int out_size, void* d_ws, size_t ws_size,
                              hipStream_t stream) {
    const float* embeds   = (const float*)d_in[0];   // [N,64]
    const float* ew       = (const float*)d_in[1];   // [E]
    const float* Wgc      = (const float*)d_in[2];   // [3,64,64]
    const float* bgc      = (const float*)d_in[3];   // [3,1,64]
    const float* Wbi      = (const float*)d_in[4];   // [3,64,64]
    const float* bbi      = (const float*)d_in[5];   // [3,1,64]
    const int*   eidx     = (const int*)d_in[6];     // [2,E]
    const int*   users    = (const int*)d_in[7];     // [4096]
    const int*   pos      = (const int*)d_in[8];
    const int*   neg      = (const int*)d_in[9];
    float*       out      = (float*)d_out;

    // Workspace layout (floats):
    //   concat [N,256] | side [N,64] | egoA [N,64] | egoB [N,64] | packedW (bf16)
    float* concat = (float*)d_ws;
    float* side   = concat + (size_t)N_NODES * 4 * DIM;
    float* egoA   = side   + (size_t)N_NODES * DIM;
    float* egoB   = egoA   + (size_t)N_NODES * DIM;
    __bf16* packedW = (__bf16*)(egoB + (size_t)N_NODES * DIM);

    // Prepack weights into bf16 WMMA fragment order (tiny: 48 KB total).
    ngcf_pack_weights_kernel<<<(N_LAYERS * PACK_PER_LAYER + 255) / 256, 256, 0, stream>>>(
        Wgc, Wbi, packedW);

    // Layer-0 slice of concat.
    ngcf_copy_embeds_kernel<<<(N_NODES * 16 + 255) / 256, 256, 0, stream>>>(embeds, concat);

    const float* ego_cur = embeds;
    float* ego_bufs[2] = { egoA, egoB };
    const int sideElems = N_NODES * DIM;

    for (int k = 0; k < N_LAYERS; ++k) {
        float* ego_next = ego_bufs[k & 1];
        ngcf_zero_kernel<<<(sideElems + 255) / 256, 256, 0, stream>>>(side, sideElems);
        ngcf_scatter_kernel<<<(N_EDGES * 16 + 255) / 256, 256, 0, stream>>>(
            ego_cur, ew, eidx, side);
        ngcf_gemm_kernel<<<N_NODES / 16, 128, 0, stream>>>(
            side, ego_cur,
            (const v16bf*)(packedW + (size_t)k * PACK_PER_LAYER),
            bgc + (size_t)k * DIM, bbi + (size_t)k * DIM,
            ego_next, concat, k);
        ego_cur = ego_next;
    }

    ngcf_gather_kernel<<<3 * BATCH, 64, 0, stream>>>(concat, users, pos, neg, out);
}
